// MultiHeadAttention_34239479284338
// MI455X (gfx1250) — matrix operand order repaired and checked by reading, not compiled, not run on hardware
//
#include <hip/hip_runtime.h>

// ---------- types ----------
typedef __attribute__((ext_vector_type(16))) __bf16        bf16x16;
typedef __attribute__((ext_vector_type(8)))  float         f32x8;
typedef __attribute__((ext_vector_type(4)))  float         f32x4;
typedef __attribute__((ext_vector_type(4)))  unsigned int  u32x4;

union FragBF {
    bf16x16 bf;
    u32x4   u[2];
};

// ---------- problem constants ----------
#define BATCH   2
#define S_LEN   2048
#define DMODEL  1024
#define NHEAD   16
#define HDIM    64
#define MAXREL  64

// ---------- gfx1250 async global->LDS copy (ASYNCcnt-tracked) ----------
__device__ __forceinline__ void async_b128(unsigned lds_off, const void* gaddr) {
    asm volatile("global_load_async_to_lds_b128 %0, %1, off"
                 :: "v"(lds_off), "v"(gaddr) : "memory");
}
__device__ __forceinline__ void wait_async0() {
    asm volatile("s_wait_asynccnt 0x0" ::: "memory");
}

// ================================================================
// Kernel 1: projection GEMM  out_bf16[b,h,s,d] = (X @ W + bias)
//   X: [B*S, D] f32,  W: [D, D] f32 row-major, out: bf16 [B,H,S,64]
// Tiling: 128x64 per 256-thread block (8 waves x 16 rows x 64 cols), BK=32
// ================================================================
#define PBM 128
#define PBN 64
#define PBK 32
#define PLA 40   // a_lds row stride (bf16 elems), 80B = 16B multiple
#define PLB 40   // bt_lds row stride

__global__ __launch_bounds__(256)
void qkv_proj_kernel(const float* __restrict__ X, const float* __restrict__ W,
                     const float* __restrict__ bias, unsigned short* __restrict__ out_raw)
{
    __bf16* __restrict__ out = reinterpret_cast<__bf16*>(out_raw);

    __shared__ __align__(16) __bf16 a_lds[PBM][PLA];   // A tile, row-major [m][k]
    __shared__ __align__(16) __bf16 bt_lds[PBN][PLB];  // B tile transposed [n][k]

    const int tid  = threadIdx.x;
    const int lane = tid & 31;
    const int wave = tid >> 5;        // 0..7, each owns 16 rows x 64 cols
    const int hi   = lane >> 4;       // 0/1
    const int ln15 = lane & 15;

    const int mb = blockIdx.y * PBM;
    const int nb = blockIdx.x * PBN;

    f32x8 acc[4];
#pragma unroll
    for (int i = 0; i < 4; i++) acc[i] = f32x8{0.f,0.f,0.f,0.f,0.f,0.f,0.f,0.f};

    for (int kb = 0; kb < DMODEL; kb += PBK) {
        __syncthreads();
        // ---- A tile: 128x32 f32 -> bf16 LDS. 256 threads * 16 elems ----
        {
            const int r  = tid >> 1;
            const int c0 = (tid & 1) * 16;
            const float* src = X + (size_t)(mb + r) * DMODEL + kb + c0;
            f32x4 v0 = *(const f32x4*)(src + 0);
            f32x4 v1 = *(const f32x4*)(src + 4);
            f32x4 v2 = *(const f32x4*)(src + 8);
            f32x4 v3 = *(const f32x4*)(src + 12);
            __bf16* dst = &a_lds[r][c0];
#pragma unroll
            for (int i = 0; i < 4; i++) {
                dst[0 + i]  = (__bf16)v0[i];
                dst[4 + i]  = (__bf16)v1[i];
                dst[8 + i]  = (__bf16)v2[i];
                dst[12 + i] = (__bf16)v3[i];
            }
        }
        // ---- B tile: W[kb..kb+31][nb..nb+63] stored transposed [n][k] ----
        {
            const int kr  = tid >> 3;        // 0..31
            const int nc0 = (tid & 7) * 8;   // 0..56
            const float* src = W + (size_t)(kb + kr) * DMODEL + nb + nc0;
#pragma unroll
            for (int i = 0; i < 8; i++)
                bt_lds[nc0 + i][kr] = (__bf16)src[i];
        }
        if (kb + PBK < DMODEL) {
            __builtin_prefetch(X + (size_t)(mb + (tid >> 1)) * DMODEL + kb + PBK, 0, 3);
            __builtin_prefetch(W + (size_t)(kb + PBK + (tid >> 3)) * DMODEL + nb, 0, 3);
        }
        __syncthreads();

        // ---- compute: batch all fragment loads, then 4 back-to-back WMMAs ----
        FragBF fa, fb[4];
        {
            const __bf16* ap = &a_lds[wave * 16 + ln15][hi * 16];
            fa.u[0] = *(const u32x4*)(ap);
            fa.u[1] = *(const u32x4*)(ap + 8);
        }
#pragma unroll
        for (int nt = 0; nt < 4; nt++) {
            const __bf16* bp = &bt_lds[nt * 16 + ln15][hi * 16];
            fb[nt].u[0] = *(const u32x4*)(bp);
            fb[nt].u[1] = *(const u32x4*)(bp + 8);
        }
#pragma unroll
        for (int nt = 0; nt < 4; nt++)
            acc[nt] = __builtin_amdgcn_wmma_f32_16x16x32_bf16(
                false, fa.bf, false, fb[nt].bf, (short)0, acc[nt], false, false);
    }

    // ---- epilogue: +bias, cvt bf16, store [B,H,S,64] ----
#pragma unroll
    for (int nt = 0; nt < 4; nt++) {
        const int n  = nb + nt * 16 + ln15;
        const float bv = bias[n];
        const int h  = n >> 6;
        const int dd = n & 63;
#pragma unroll
        for (int g = 0; g < 8; g++) {
            const int mrow = mb + wave * 16 + g + hi * 8;
            const int bix  = mrow >> 11;        // / 2048
            const int s    = mrow & 2047;
            const size_t off = ((size_t)(bix * NHEAD + h) * S_LEN + s) * HDIM + dd;
            out[off] = (__bf16)(acc[nt][g] + bv);
        }
    }
}

// ================================================================
// Kernel 2: fused flash attention with relative-position key bias
//   grid (S/64, H, B), 128 threads (4 waves x 16 q-rows each)
//   K/V/mask double-buffered in LDS via GLOBAL_LOAD_ASYNC_TO_LDS_B128
// ================================================================
#define ABM 64
#define ABN 64
#define QSTR 72
#define KSTR 72
#define VSTR 72
#define PSTR 72
#define RSTR 148
#define TSTR 72
#define NKB  (S_LEN / ABN)

__global__ __launch_bounds__(128)
void attn_kernel(const unsigned short* __restrict__ q_raw,
                 const unsigned short* __restrict__ k_raw,
                 const unsigned short* __restrict__ v_raw,
                 const float* __restrict__ mask,
                 const float* __restrict__ rel,
                 float* __restrict__ out)
{
    const __bf16* __restrict__ qws = reinterpret_cast<const __bf16*>(q_raw);
    const __bf16* __restrict__ kws = reinterpret_cast<const __bf16*>(k_raw);
    const __bf16* __restrict__ vws = reinterpret_cast<const __bf16*>(v_raw);

    __shared__ __align__(16) __bf16 q_lds[ABM][QSTR];        // [qrow][d]
    __shared__ __align__(16) __bf16 k_lds[2][ABN][KSTR];     // [buf][key][d]  (= B^T for QK^T)
    __shared__ __align__(16) __bf16 v_stage[2][ABN][VSTR];   // [buf][key][d]  async landing pad
    __shared__ __align__(16) __bf16 vt_lds[HDIM][VSTR];      // [d][key]       (= B^T for PV)
    __shared__ __align__(16) __bf16 relt[144][TSTR];         // [t][d]         (= B^T for Q·relT)
    __shared__ __align__(16) __bf16 p_lds[4][16][PSTR];      // per-wave P tile [m][key]
    __shared__ float r_lds[ABM][RSTR];                       // r[qrow][t], t in 0..128
    __shared__ __align__(16) float mask_lds[2][ABN];

    const int tid  = threadIdx.x;
    const int lane = tid & 31;
    const int wave = tid >> 5;      // 0..3
    const int hi   = lane >> 4;
    const int ln15 = lane & 15;

    const int b  = blockIdx.z;
    const int h  = blockIdx.y;
    const int qb = blockIdx.x * ABM;

    const size_t headoff = (size_t)(b * NHEAD + h) * S_LEN * HDIM;

    // issue async copies for K/V/mask of block kb into buffer `buf`
    auto issue_kv = [&](int kb, int buf) {
        // 64 rows x 128B each = 512 x 16B chunks; 128 threads x 4 chunks
#pragma unroll
        for (int i = 0; i < 4; i++) {
            const int c   = tid + i * 128;
            const int row = c >> 3;
            const int col = (c & 7) * 8;
            const __bf16* gk = kws + headoff + (size_t)(kb + row) * HDIM + col;
            const __bf16* gv = vws + headoff + (size_t)(kb + row) * HDIM + col;
            async_b128((unsigned)(size_t)&k_lds[buf][row][col], gk);
            async_b128((unsigned)(size_t)&v_stage[buf][row][col], gv);
        }
        if (tid < 16)
            async_b128((unsigned)(size_t)&mask_lds[buf][tid * 4],
                       mask + b * S_LEN + kb + tid * 4);
    };

    // ---- async-load Q tile (64x64 bf16) ----
#pragma unroll
    for (int i = 0; i < 4; i++) {
        const int c   = tid + i * 128;
        const int row = c >> 3;
        const int col = (c & 7) * 8;
        async_b128((unsigned)(size_t)&q_lds[row][col],
                   qws + headoff + (size_t)(qb + row) * HDIM + col);
    }
    // ---- prefetch first K/V/mask block ----
    issue_kv(0, 0);

    // ---- load rel_table (129x64 f32 -> bf16, zero-pad to 144 rows) ----
    for (int idx = tid; idx < 144 * 64; idx += 128) {
        const int t = idx >> 6, d = idx & 63;
        const float v = (t < 129) ? rel[t * HDIM + d] : 0.0f;
        relt[t][d] = (__bf16)v;
    }

    wait_async0();
    __syncthreads();

    // ---- Q A-fragments: invariant across all K blocks -> hoist to registers ----
    FragBF faq[2];
#pragma unroll
    for (int kk = 0; kk < 2; kk++) {
        const __bf16* ap = &q_lds[wave * 16 + ln15][kk * 32 + hi * 16];
        faq[kk].u[0] = *(const u32x4*)ap;
        faq[kk].u[1] = *(const u32x4*)(ap + 8);
    }

    // ---- r[q,t] = Q @ rel^T : 9 n-tiles x 2 k-steps of WMMA per wave ----
#pragma unroll
    for (int nt = 0; nt < 9; nt++) {
        f32x8 racc = f32x8{0.f,0.f,0.f,0.f,0.f,0.f,0.f,0.f};
#pragma unroll
        for (int kk = 0; kk < 2; kk++) {
            FragBF fb;
            const __bf16* bp = &relt[nt * 16 + ln15][kk * 32 + hi * 16];
            fb.u[0] = *(const u32x4*)bp;
            fb.u[1] = *(const u32x4*)(bp + 8);
            racc = __builtin_amdgcn_wmma_f32_16x16x32_bf16(
                false, faq[kk].bf, false, fb.bf, (short)0, racc, false, false);
        }
#pragma unroll
        for (int g = 0; g < 8; g++)
            r_lds[wave * 16 + g + hi * 8][nt * 16 + ln15] = racc[g];
    }
    // (each wave only reads back its own r rows -> no cross-wave barrier needed)

    // ---- flash-attention state ----
    float m_st[8], l_st[8];
    f32x8 ctx[4];
#pragma unroll
    for (int g = 0; g < 8; g++) { m_st[g] = -1e30f; l_st[g] = 0.0f; }
#pragma unroll
    for (int i = 0; i < 4; i++) ctx[i] = f32x8{0.f,0.f,0.f,0.f,0.f,0.f,0.f,0.f};

    const float scale = 0.125f;   // 1/sqrt(64)
    const float L2E   = 1.44269504088896340736f;
    const int   qrow0 = qb + wave * 16;

    for (int ib = 0; ib < NKB; ib++) {
        const int kb  = ib * ABN;
        const int buf = ib & 1;

        if (ib) {               // buf's async copies were issued last iteration
            wait_async0();
            __syncthreads();
        }
        if (ib + 1 < NKB) issue_kv(kb + ABN, buf ^ 1);   // prefetch next block

        // ---- transpose V stage [key][d] -> vt [d][key] (LDS->LDS) ----
        {
            const int r  = tid >> 1;
            const int c0 = (tid & 1) * 32;
            union { u32x4 u[4]; __bf16 e[32]; } vv;
            const u32x4* s4 = (const u32x4*)&v_stage[buf][r][c0];
            vv.u[0] = s4[0]; vv.u[1] = s4[1]; vv.u[2] = s4[2]; vv.u[3] = s4[3];
#pragma unroll
            for (int i = 0; i < 32; i++) vt_lds[c0 + i][r] = vv.e[i];
        }
        __syncthreads();

        // ---- scores = Q K^T : Q frags in regs, 8 B-frags / WMMAs ----
        f32x8 sacc[4];
#pragma unroll
        for (int nt = 0; nt < 4; nt++)
            sacc[nt] = f32x8{0.f,0.f,0.f,0.f,0.f,0.f,0.f,0.f};
#pragma unroll
        for (int kk = 0; kk < 2; kk++) {
#pragma unroll
            for (int nt = 0; nt < 4; nt++) {
                FragBF fb;
                const __bf16* bp = &k_lds[buf][nt * 16 + ln15][kk * 32 + hi * 16];
                fb.u[0] = *(const u32x4*)bp;
                fb.u[1] = *(const u32x4*)(bp + 8);
                sacc[nt] = __builtin_amdgcn_wmma_f32_16x16x32_bf16(
                    false, faq[kk].bf, false, fb.bf, (short)0, sacc[nt], false, false);
            }
        }

        // ---- + rel bias (gather from r_lds), *scale, + mask ----
#pragma unroll
        for (int nt = 0; nt < 4; nt++) {
            const int klo  = kb + nt * 16;
            const float mv = mask_lds[buf][nt * 16 + ln15];
            if (klo - (qrow0 + 15) >= MAXREL) {
                // fully clipped high: dist=+64 -> idx 128, constant per row
#pragma unroll
                for (int g = 0; g < 8; g++) {
                    const int row = wave * 16 + g + hi * 8;
                    sacc[nt][g] = (sacc[nt][g] + r_lds[row][2 * MAXREL]) * scale + mv;
                }
            } else if (qrow0 - (klo + 15) >= MAXREL) {
                // fully clipped low: dist=-64 -> idx 0
#pragma unroll
                for (int g = 0; g < 8; g++) {
                    const int row = wave * 16 + g + hi * 8;
                    sacc[nt][g] = (sacc[nt][g] + r_lds[row][0]) * scale + mv;
                }
            } else {
                const int kcol = klo + ln15;
#pragma unroll
                for (int g = 0; g < 8; g++) {
                    const int row = wave * 16 + g + hi * 8;
                    int d = kcol - (qb + row);
                    d = d < -MAXREL ? -MAXREL : (d > MAXREL ? MAXREL : d);
                    sacc[nt][g] = (sacc[nt][g] + r_lds[row][d + MAXREL]) * scale + mv;
                }
            }
        }

        // ---- online softmax (rows live across 16-lane halves) ----
#pragma unroll
        for (int g = 0; g < 8; g++) {
            float rm = fmaxf(fmaxf(sacc[0][g], sacc[1][g]), fmaxf(sacc[2][g], sacc[3][g]));
#pragma unroll
            for (int off = 8; off >= 1; off >>= 1) rm = fmaxf(rm, __shfl_xor(rm, off, 16));
            const float mnew = fmaxf(m_st[g], rm);
            const float f = exp2f((m_st[g] - mnew) * L2E);
            m_st[g] = mnew;
            float rs = 0.0f;
#pragma unroll
            for (int nt = 0; nt < 4; nt++) {
                const float p = exp2f((sacc[nt][g] - mnew) * L2E);
                sacc[nt][g] = p;
                rs += p;
            }
#pragma unroll
            for (int off = 8; off >= 1; off >>= 1) rs += __shfl_xor(rs, off, 16);
            l_st[g] = l_st[g] * f + rs;
#pragma unroll
            for (int dt = 0; dt < 4; dt++) ctx[dt][g] *= f;
        }

        // ---- P -> per-wave LDS (re-fragment as A) ----
#pragma unroll
        for (int nt = 0; nt < 4; nt++)
#pragma unroll
            for (int g = 0; g < 8; g++)
                p_lds[wave][g + hi * 8][nt * 16 + ln15] = (__bf16)sacc[nt][g];

        // ---- ctx += P @ V : A-frag per k-step, 8 B-frags / WMMAs ----
#pragma unroll
        for (int kk = 0; kk < 2; kk++) {
            FragBF fa;
            const __bf16* ap = &p_lds[wave][ln15][kk * 32 + hi * 16];
            fa.u[0] = *(const u32x4*)ap;
            fa.u[1] = *(const u32x4*)(ap + 8);
#pragma unroll
            for (int dt = 0; dt < 4; dt++) {
                FragBF fb;
                const __bf16* bp = &vt_lds[dt * 16 + ln15][kk * 32 + hi * 16];
                fb.u[0] = *(const u32x4*)bp;
                fb.u[1] = *(const u32x4*)(bp + 8);
                ctx[dt] = __builtin_amdgcn_wmma_f32_16x16x32_bf16(
                    false, fa.bf, false, fb.bf, (short)0, ctx[dt], false, false);
            }
        }
    }

    // ---- epilogue: normalize by l, store f32 [B,S,D] ----
#pragma unroll
    for (int g = 0; g < 8; g++) {
        const float inv = 1.0f / l_st[g];
        const int s = qb + wave * 16 + g + hi * 8;
#pragma unroll
        for (int dt = 0; dt < 4; dt++) {
            const int dd = dt * 16 + ln15;
            out[(size_t)(b * S_LEN + s) * DMODEL + h * HDIM + dd] = ctx[dt][g] * inv;
        }
    }
}

// ================================================================
// launcher
// ================================================================
extern "C" void kernel_launch(void* const* d_in, const int* in_sizes, int n_in,
                              void* d_out, int out_size, void* d_ws, size_t ws_size,
                              hipStream_t stream)
{
    const float* query = (const float*)d_in[0];
    const float* key   = (const float*)d_in[1];
    const float* value = (const float*)d_in[2];
    const float* amask = (const float*)d_in[3];
    const float* Wq    = (const float*)d_in[4];
    const float* bq    = (const float*)d_in[5];
    const float* Wk    = (const float*)d_in[6];
    const float* bk    = (const float*)d_in[7];
    const float* Wv    = (const float*)d_in[8];
    const float* bv    = (const float*)d_in[9];
    const float* rel   = (const float*)d_in[10];

    const size_t elems = (size_t)BATCH * S_LEN * DMODEL;   // 4M bf16 each
    unsigned short* qbf = (unsigned short*)d_ws;
    unsigned short* kbf = qbf + elems;
    unsigned short* vbf = kbf + elems;

    dim3 pgrid(DMODEL / PBN, (BATCH * S_LEN) / PBM);   // (16, 32)
    qkv_proj_kernel<<<pgrid, 256, 0, stream>>>(query, Wq, bq, qbf);
    qkv_proj_kernel<<<pgrid, 256, 0, stream>>>(key,   Wk, bk, kbf);
    qkv_proj_kernel<<<pgrid, 256, 0, stream>>>(value, Wv, bv, vbf);

    dim3 agrid(S_LEN / ABM, NHEAD, BATCH);             // (32, 16, 2)
    attn_kernel<<<agrid, 128, 0, stream>>>(qbf, kbf, vbf, amask, rel, (float*)d_out);
}
